// EncoderBlock_79353815761494
// MI455X (gfx1250) — matrix operand order repaired and checked by reading, not compiled, not run on hardware
//
#include <hip/hip_runtime.h>
#include <hip/hip_bf16.h>
#include <stdint.h>

// ---------------------------------------------------------------------------
// Problem constants (match the JAX reference)
// ---------------------------------------------------------------------------
#define NUM_COLS 4
#define NUM_ROWS 4
#define CIN 16            // per-stream channels
#define BATCH 8
#define HH 128
#define WW 128
#define NCH 48            // 3*CIN conv in/out channels
#define TAPS 9
#define KTOT (TAPS * NCH)           // 432 true K extent
#define NKCHUNK 14                  // ceil(432/32) K=32 WMMA steps
#define WROW 448                    // packed weight row: 432 real + 16 zero halves
#define STRIDE 56                   // LDS pixel stride in halves (112 B, 16B aligned)
#define PATCH_W 130                 // w = -1 .. 128
#define PATCH_ELEMS (3 * PATCH_W * STRIDE + 64)   // +64 halves slack for tail chunk

typedef __bf16 bf16_t;
typedef __attribute__((ext_vector_type(16))) __bf16 bf16x16;
typedef __attribute__((ext_vector_type(8)))  float  f32x8;

// Compile-time A-group offset for K position k (multiple of 8):
//   tap = k/48 (clamped to 8 for the zero-weighted tail), c = k - tap*48
//   offset = (dh*PATCH_W + dw)*STRIDE + c
// Note: group starts are multiples of 16, so aoff(k+8) == aoff(k)+8 always;
// the lane-dependent part is folded into the base pointer (hiHalf*8).
__host__ __device__ constexpr int aoff(int k) {
    int tap = k / NCH;
    if (tap > TAPS - 1) tap = TAPS - 1;     // tail: garbage reads x zero weights
    int c  = k - tap * NCH;
    int dh = tap / 3;
    int dw = tap - dh * 3;
    return (dh * PATCH_W + dw) * STRIDE + c;
}

// ---------------------------------------------------------------------------
// Pack weights: W fp32 [col][row][48][48][3][3] -> bf16 [cell][n][448]
// row layout: k = tap*48 + c for k<432, zeros for k in [432,448).
// ---------------------------------------------------------------------------
__global__ void pack_weights_kernel(const float* __restrict__ Wsrc,
                                    bf16_t* __restrict__ Wdst, int total) {
    int i = blockIdx.x * blockDim.x + threadIdx.x;
    if (i >= total) return;
    int kk   =  i % WROW;
    int n    = (i / WROW) % NCH;
    int cell =  i / (WROW * NCH);
    float v = 0.0f;
    if (kk < KTOT) {
        int tap = kk / NCH;
        int c   = kk - tap * NCH;
        // Wsrc flat: (((cell*48 + n)*48 + c)*3 + kh)*3 + kw, tap = kh*3+kw
        v = Wsrc[((size_t)(cell * NCH + n) * NCH + c) * TAPS + tap];
    }
    Wdst[i] = (bf16_t)v;
}

// ---------------------------------------------------------------------------
// Pack input: x fp32 [B][16][H][W] -> bf16 stream [B][H][W][16]
// ---------------------------------------------------------------------------
__global__ void pack_input_kernel(const float* __restrict__ x,
                                  bf16_t* __restrict__ dst, int total) {
    int i = blockIdx.x * blockDim.x + threadIdx.x;
    if (i >= total) return;
    int c =  i & (CIN - 1);
    int w = (i >> 4) & (WW - 1);
    int h = (i / (CIN * WW)) & (HH - 1);
    int b =  i / (CIN * WW * HH);
    dst[i] = (bf16_t)x[(((size_t)b * CIN + c) * HH + h) * WW + w];
}

// ---------------------------------------------------------------------------
// Zero-fill a bf16 stream buffer (used for missing input slots)
// ---------------------------------------------------------------------------
__global__ void zero_stream_kernel(uint32_t* __restrict__ p, int n_u32) {
    int i = blockIdx.x * blockDim.x + threadIdx.x;
    if (i < n_u32) p[i] = 0u;
}

// ---------------------------------------------------------------------------
// One wavefront-cell conv: 3x3, 48->48 ch, SAME pad, bias + ReLU.
// Implicit GEMM, unpadded K = 9*48 = 432 -> 14 v_wmma_f32_16x16x32_bf16 steps.
//   grid  = B*H blocks (one image row each), block = 256 threads = 8 wave32
//   wave w: M-tile = pixels [w*16, w*16+16), N = all 48 output channels
// ---------------------------------------------------------------------------
__global__ __launch_bounds__(256)
void conv_cell_kernel(const bf16_t* __restrict__ s0,
                      const bf16_t* __restrict__ s1,
                      const bf16_t* __restrict__ s2,
                      const bf16_t* __restrict__ wcell,   // [48][448]
                      const float*  __restrict__ bias,    // [48]
                      bf16_t* __restrict__ outR,
                      bf16_t* __restrict__ outU,
                      bf16_t* __restrict__ outD) {
    __shared__ bf16_t patch[PATCH_ELEMS];               // ~43.8 KB

    const int tid  = threadIdx.x;
    const int lane = tid & 31;
    const int wave = tid >> 5;           // 0..7
    const int bh   = blockIdx.x;         // b*H + h
    const int b    = bh >> 7;            // /128
    const int h    = bh & (HH - 1);

    // Warm the (L2-resident) packed weight panel for this cell.
    __builtin_prefetch(wcell + (size_t)tid * 168, 0, 1);   // 48*448*2B / 256thr

    // ---- Stage 3x130 pixel patch (48 ch, 3 slots of 16) into LDS ----
    const int NSTAGE = 3 * PATCH_W * 3;
    for (int i = tid; i < NSTAGE; i += 256) {
        const int slot = i % 3;
        const int pix  = (i / 3) % PATCH_W;
        const int r    = (i / 3) / PATCH_W;
        const int hs   = h + r - 1;
        const int ws   = pix - 1;
        uint4 lo = make_uint4(0u, 0u, 0u, 0u);
        uint4 hi = make_uint4(0u, 0u, 0u, 0u);
        if (hs >= 0 && hs < HH && ws >= 0 && ws < WW) {
            const bf16_t* sp = (slot == 0) ? s0 : ((slot == 1) ? s1 : s2);
            const uint4* src =
                (const uint4*)(sp + ((size_t)(b * HH + hs) * WW + ws) * CIN);
            lo = src[0];
            hi = src[1];
        }
        uint4* dst = (uint4*)(&patch[(r * PATCH_W + pix) * STRIDE + slot * 16]);
        dst[0] = lo;
        dst[1] = hi;
    }
    __syncthreads();

    // ---- GEMM: M=16 pixels, N=48 (3 tiles), K=432 over 14 K=32 chunks ----
    const int nlane  = lane & 15;        // A: row M, B/C: col N
    const int hiHalf = lane >> 4;        // lane group 0/1
    const int wbase  = wave * 16;

    // Single per-lane, loop-invariant A base; every A-load offset is then a compile-time
    // immediate on the ds_load (aoff(k+8) == aoff(k)+8 for all groups).
    const bf16_t* abase = &patch[(wbase + nlane) * STRIDE + hiHalf * 8];
    // Single per-lane, loop-invariant B base; per-chunk / per-tile offsets fold into
    // global_load immediates.
    const bf16_t* bbase = wcell + (size_t)nlane * WROW + hiHalf * 8;

    f32x8 acc[3];
    acc[0] = (f32x8)0.0f; acc[1] = (f32x8)0.0f; acc[2] = (f32x8)0.0f;

    union AF { bf16x16 v; uint4 q[2]; };
    union BF { bf16x16 v; uint4 q[2]; };

#pragma unroll
    for (int j = 0; j < NKCHUNK; ++j) {
        const int k0 = 32 * j;
        // A fragment: lanes 0-15 hold K k0+0..7 & k0+16..23,
        //             lanes 16-31 hold K k0+8..15 & k0+24..31.
        AF a;
        a.q[0] = *(const uint4*)(abase + aoff(k0));
        a.q[1] = *(const uint4*)(abase + aoff(k0 + 16));
#pragma unroll
        for (int t = 0; t < 3; ++t) {
            // B fragment: same K order as A (k0+8h+0..7 & k0+16+8h+0..7).
            const bf16_t* wp = bbase + (size_t)(t * 16) * WROW + k0;
            BF bf;
            bf.q[0] = ((const uint4*)wp)[0];
            bf.q[1] = ((const uint4*)wp)[2];
            acc[t] = __builtin_amdgcn_wmma_f32_16x16x32_bf16(
                false, a.v, false, bf.v, (short)0, acc[t], false, false);
        }
    }

    // ---- Epilogue: bias + ReLU, scatter to right/up/down bf16 streams ----
    bf16_t* outs[3] = { outR, outU, outD };
    const size_t rowbase = (size_t)(b * HH + h) * WW;
#pragma unroll
    for (int t = 0; t < 3; ++t) {
        const float bv = bias[t * 16 + nlane];
        bf16_t* op = outs[t] + rowbase * CIN + nlane;
#pragma unroll
        for (int r = 0; r < 8; ++r) {
            const int m = r + hiHalf * 8;            // C layout: lanes16-31 -> M+8
            float v = acc[t][r] + bv;
            v = v > 0.0f ? v : 0.0f;
            op[(size_t)(wbase + m) * CIN] = (bf16_t)v;
        }
    }
}

// ---------------------------------------------------------------------------
// Export: final right-stream bf16 [B][H][W][16] -> fp32 NCHW d_out
// ---------------------------------------------------------------------------
__global__ void export_kernel(const bf16_t* __restrict__ rs,
                              float* __restrict__ out, int total) {
    int i = blockIdx.x * blockDim.x + threadIdx.x;
    if (i >= total) return;
    int w =  i & (WW - 1);
    int h = (i / WW) & (HH - 1);
    int c = (i / (WW * HH)) & (CIN - 1);
    int b =  i / (WW * HH * CIN);
    out[i] = (float)rs[((size_t)(b * HH + h) * WW + w) * CIN + c];
}

// ---------------------------------------------------------------------------
// Host-side orchestration
// ---------------------------------------------------------------------------
extern "C" void kernel_launch(void* const* d_in, const int* in_sizes, int n_in,
                              void* d_out, int out_size, void* d_ws, size_t ws_size,
                              hipStream_t stream) {
    (void)in_sizes; (void)n_in; (void)ws_size;
    const float* x  = (const float*)d_in[0];  // [8,16,128,128]
    const float* Wf = (const float*)d_in[1];  // [4,4,48,48,3,3]
    const float* bf = (const float*)d_in[2];  // [4,4,48]
    float* out = (float*)d_out;

    const size_t STREAM_ELEMS = (size_t)BATCH * HH * WW * CIN;     // 2,097,152
    const size_t STREAM_BYTES = STREAM_ELEMS * sizeof(bf16_t);     // 4 MB
    const size_t WPACK_ELEMS  = (size_t)NUM_COLS * NUM_ROWS * NCH * WROW;
    const size_t WPACK_BYTES  = (WPACK_ELEMS * sizeof(bf16_t) + 255) & ~(size_t)255;

    char* ws = (char*)d_ws;
    bf16_t* wpack = (bf16_t*)ws;                 ws += WPACK_BYTES;
    bf16_t* inStream = (bf16_t*)ws;              ws += STREAM_BYTES;
    bf16_t* zeroBuf  = (bf16_t*)ws;              ws += STREAM_BYTES;
    bf16_t* rightB[NUM_ROWS][2];
    bf16_t* upB[NUM_ROWS];
    bf16_t* downB[NUM_ROWS];
    for (int r = 0; r < NUM_ROWS; ++r)
        for (int p = 0; p < 2; ++p) { rightB[r][p] = (bf16_t*)ws; ws += STREAM_BYTES; }
    for (int r = 0; r < NUM_ROWS; ++r) { upB[r]   = (bf16_t*)ws; ws += STREAM_BYTES; }
    for (int r = 0; r < NUM_ROWS; ++r) { downB[r] = (bf16_t*)ws; ws += STREAM_BYTES; }

    // Prep: pack weights / input, zero the filler stream.
    {
        int n = (int)WPACK_ELEMS;
        pack_weights_kernel<<<(n + 255) / 256, 256, 0, stream>>>(Wf, wpack, n);
    }
    {
        int n = (int)STREAM_ELEMS;
        pack_input_kernel<<<(n + 255) / 256, 256, 0, stream>>>(x, inStream, n);
        int nz = (int)(STREAM_BYTES / 4);
        zero_stream_kernel<<<(nz + 255) / 256, 256, 0, stream>>>((uint32_t*)zeroBuf, nz);
    }

    // Wavefront sweep, sequential launches = dependency order.
    const int gridConv = BATCH * HH;            // 1024 blocks
    for (int col = 0; col < NUM_COLS; ++col) {
        for (int row = 0; row < NUM_ROWS; ++row) {
            const int cell = col * NUM_ROWS + row;
            const bf16_t *s0, *s1, *s2;
            if (col == 0) {
                s0 = (row == 0) ? inStream : downB[row - 1];
                s1 = zeroBuf; s2 = zeroBuf;
            } else {
                s0 = rightB[row][(col - 1) & 1];
                if (row == 0)               { s1 = upB[1];          s2 = zeroBuf; }
                else if (row == NUM_ROWS-1) { s1 = downB[row - 1];  s2 = zeroBuf; }
                else                        { s1 = upB[row + 1];    s2 = downB[row - 1]; }
            }
            conv_cell_kernel<<<gridConv, 256, 0, stream>>>(
                s0, s1, s2,
                wpack + (size_t)cell * NCH * WROW,
                bf + (size_t)cell * NCH,
                rightB[row][col & 1], upB[row], downB[row]);
        }
    }

    // Final output = right stream of cell (col=3,row=3).
    export_kernel<<<(out_size + 255) / 256, 256, 0, stream>>>(
        rightB[NUM_ROWS - 1][(NUM_COLS - 1) & 1], out, out_size);
}
